// MambaDehaze_66889820668462
// MI455X (gfx1250) — matrix operand order repaired and checked by reading, not compiled, not run on hardware
//
#include <hip/hip_runtime.h>
#include <hip/hip_bf16.h>
#include <hip/hip_fp16.h>
#include <math.h>

// ---------------------------------------------------------------------------
// MambaDehaze on MI455X (gfx1250, wave32, WMMA).
//   * All GEMMs: v_wmma_f32_16x16x32_f16, f16 A and packed f16 B[N][K],
//     double-buffered LDS tiles staged with GLOBAL_LOAD_ASYNC_TO_LDS_B128
//     (ASYNCcnt) when available, else b128 load + b128 ds_store.
//   * SSM scan is the rank-16 recurrence (not the O(L^2) conv kernel).
//   * Bicubic upsample is 4-tap separable, fused with the smooth 3x3 conv.
// ---------------------------------------------------------------------------

typedef __attribute__((ext_vector_type(16))) _Float16 v16h;
typedef __attribute__((ext_vector_type(8)))  _Float16 v8h;
typedef __attribute__((ext_vector_type(8)))  float    v8f;
typedef __attribute__((ext_vector_type(4)))  int      v4i;

#define GASP __attribute__((address_space(1)))   // global
#define LASP __attribute__((address_space(3)))   // LDS

#define DM   128          // D_MODEL
#define LSEQ 1024         // tokens per image (32x32 patches)
#define MROWS 8192        // BATCH * LSEQ

#if __has_builtin(__builtin_amdgcn_global_load_async_to_lds_b128) && \
    __has_builtin(__builtin_amdgcn_s_wait_asynccnt)
#define USE_ASYNC_LDS 1
#else
#define USE_ASYNC_LDS 0
#endif

__device__ __forceinline__ float gelu_f(float v)    { return 0.5f * v * (1.f + erff(v * 0.70710678118f)); }
__device__ __forceinline__ float sigmoid_f(float v) { return 1.f / (1.f + __expf(-v)); }

// ---------------------------------------------------------------------------
// Generic WMMA GEMM: C[8192 x N] = A[8192 x K] * Bp^T  (+ epilogues)
//   A  : f16 [M][K] row-major          (lda = K stride)
//   Bp : f16 packed [N][K] row-major
// block = 256 threads = 8 waves (4x2), block tile 64x32, wave tile 16x16,
// K-step 32 via v_wmma_f32_16x16x32_f16. All K,N used are multiples of 32.
// ---------------------------------------------------------------------------
enum { EPI_BIAS = 0, EPI_BIAS_POS = 1, EPI_BIAS_RES_CLIP = 2 };

template <int EPI, bool CHALF>
__global__ __launch_bounds__(256) void gemm_wmma_kernel(
    const _Float16* __restrict__ A, int lda,
    const _Float16* __restrict__ Bp,          // packed [N][K]
    const float* __restrict__ bias,
    const float* __restrict__ res, int ldr,
    const float* __restrict__ pos,
    void* __restrict__ Cv, int ldc, int K)
{
    __shared__ __align__(16) _Float16 As[2][64][32];   // [m][k]  2 x 4 KB
    __shared__ __align__(16) _Float16 Bs[2][32][32];   // [n][k]  2 x 2 KB

    const int tid  = threadIdx.x;
    const int lane = tid & 31;
    const int wave = tid >> 5;
    const int wm   = wave >> 1;        // 0..3  (M tiles)
    const int wn   = wave & 1;         // 0..1  (N tiles)
    const int bm   = blockIdx.x * 64;
    const int bn   = blockIdx.y * 32;
    const int h    = lane >> 4;        // half-wave select
    const int l16  = lane & 15;

    // staging ownership: thread t -> A row t>>2, 8 contiguous cols 8*(t&3)
    const int ar = tid >> 2, ac = (tid & 3) * 8;

    auto stage = [&](int k0, int buf) {
#if USE_ASYNC_LDS
        // CDNA5 async copy: global -> LDS without a VGPR round trip (ASYNCcnt)
        __builtin_amdgcn_global_load_async_to_lds_b128(
            (GASP v4i*)(A + (size_t)(bm + ar) * lda + k0 + ac),
            (LASP v4i*)&As[buf][ar][ac], 0, 0);
        if (tid < 128)
            __builtin_amdgcn_global_load_async_to_lds_b128(
                (GASP v4i*)(Bp + (size_t)(bn + (tid >> 2)) * K + k0 + (tid & 3) * 8),
                (LASP v4i*)&Bs[buf][tid >> 2][(tid & 3) * 8], 0, 0);
#else
        *(v8h*)&As[buf][ar][ac] =
            *(const v8h*)(A + (size_t)(bm + ar) * lda + k0 + ac);
        if (tid < 128)   // B tile: 32 rows(n) x 32 cols(k), threads 0..127
            *(v8h*)&Bs[buf][tid >> 2][(tid & 3) * 8] =
                *(const v8h*)(Bp + (size_t)(bn + (tid >> 2)) * K + k0 + (tid & 3) * 8);
#endif
    };

    v8f acc = {};
    stage(0, 0);
    int p = 0;
    for (int k0 = 0; k0 < K; k0 += 32) {
#if USE_ASYNC_LDS
        __builtin_amdgcn_s_wait_asynccnt(0);
#endif
        __syncthreads();
        if (k0 + 32 < K) stage(k0 + 32, p ^ 1);
        if (k0 + 64 < K) {   // L2 prefetch two tiles ahead (global_prefetch_b8)
            __builtin_prefetch(A + (size_t)(bm + ar) * lda + k0 + 64 + ac, 0, 2);
            if (tid < 128)
                __builtin_prefetch(Bp + (size_t)(bn + (tid >> 2)) * K + k0 + 64, 0, 2);
        }

        // --- fragment loads (ISA 7.12.2 16-bit layouts), ds_load_b128 ------
        // A lane(m=l16): elems 0..7 -> K = 8h..8h+7 ; 8..15 -> K = 16+8h..
        // B lane(n=l16): elems 0..7 -> K = 8h..8h+7 ; 8..15 -> K = 16+8h..
        const int ml = wm * 16 + l16;
        const int nl = wn * 16 + l16;
        v8h alo = *(const v8h*)&As[p][ml][8 * h];
        v8h ahi = *(const v8h*)&As[p][ml][16 + 8 * h];
        v8h blo = *(const v8h*)&Bs[p][nl][8 * h];
        v8h bhi = *(const v8h*)&Bs[p][nl][16 + 8 * h];
        v16h a, bf;
        #pragma unroll
        for (int i = 0; i < 8; ++i) {
            a[i] = alo[i];  a[i + 8] = ahi[i];
            bf[i] = blo[i]; bf[i + 8] = bhi[i];
        }
        acc = __builtin_amdgcn_wmma_f32_16x16x32_f16(
                  false, a, false, bf, (short)0, acc, false, false);
        p ^= 1;
    }

    // --- epilogue: C/D layout VGPR r -> m = r + 8h, n = l16 ----------------
    #pragma unroll
    for (int r = 0; r < 8; ++r) {
        int m = bm + wm * 16 + r + 8 * h;
        int n = bn + wn * 16 + l16;
        float v = acc[r] + bias[n];
        if constexpr (EPI == EPI_BIAS_POS)
            v += pos[(size_t)(m & (LSEQ - 1)) * ldc + n];
        if constexpr (EPI == EPI_BIAS_RES_CLIP) {
            v += res[(size_t)m * ldr + n];
            v = fminf(10.f, fmaxf(-10.f, v));
        }
        if constexpr (CHALF)
            ((_Float16*)Cv)[(size_t)m * ldc + n] = (_Float16)v;
        else
            ((float*)Cv)[(size_t)m * ldc + n] = v;
    }
}

// ---------------------------------------------------------------------------
// Weight packing: W[K][N] fp32 -> out[N][K] f16  (transpose + convert)
// ---------------------------------------------------------------------------
__global__ __launch_bounds__(256) void pack_wt_kernel(
    const float* __restrict__ W, _Float16* __restrict__ out, int K, int N)
{
    int idx = blockIdx.x * 256 + threadIdx.x;
    if (idx >= K * N) return;
    int n = idx / K, k = idx % K;
    out[idx] = (_Float16)W[(size_t)k * N + n];
}

// direct fp32 -> f16 convert (already [N][K] layout, e.g. sharp conv weight)
__global__ __launch_bounds__(256) void pack_direct_kernel(
    const float* __restrict__ W, _Float16* __restrict__ out, int total)
{
    int idx = blockIdx.x * 256 + threadIdx.x;
    if (idx < total) out[idx] = (_Float16)W[idx];
}

// ---------------------------------------------------------------------------
// LayerNorm over last dim (128), one row per 128-thread block.
// Writes f16 copy (GEMM A operand) and optionally fp32.
// ---------------------------------------------------------------------------
__global__ __launch_bounds__(128) void ln_kernel(
    const float* __restrict__ in, const float* __restrict__ g,
    const float* __restrict__ b, float* __restrict__ outf,
    _Float16* __restrict__ outh)
{
    __shared__ float red[128];
    const size_t row = (size_t)blockIdx.x * DM;
    const int d = threadIdx.x;
    float x = in[row + d];
    red[d] = x; __syncthreads();
    #pragma unroll
    for (int s = 64; s > 0; s >>= 1) { if (d < s) red[d] += red[d + s]; __syncthreads(); }
    float mu = red[0] * (1.f / DM);
    __syncthreads();
    float dv = x - mu;
    red[d] = dv * dv; __syncthreads();
    #pragma unroll
    for (int s = 64; s > 0; s >>= 1) { if (d < s) red[d] += red[d + s]; __syncthreads(); }
    float var = red[0] * (1.f / DM);
    float y = dv * rsqrtf(var + 1e-5f) * g[d] + b[d];
    outh[row + d] = (_Float16)y;
    if (outf) outf[row + d] = y;
}

// ---------------------------------------------------------------------------
// S4 selective-scan as a rank-16 recurrence. One block per batch image,
// thread d carries 16 states in registers; dir=+1 forward, -1 backward.
// Output (f16) y[t,d] = (C . h[t]) * silu(z[t,d]).
// ---------------------------------------------------------------------------
__global__ __launch_bounds__(128) void scan_kernel(
    const float* __restrict__ xz, const float* __restrict__ A_log,
    const float* __restrict__ Bm, const float* __restrict__ Cm,
    const float* __restrict__ log_dt, _Float16* __restrict__ y, int dir)
{
    const int b = blockIdx.x;
    const int d = threadIdx.x;
    float dt = __expf(log_dt[d]);
    dt = fminf(1.f, fmaxf(1e-4f, dt));
    float Ab[16], Bd[16], Cc[16], hst[16];
    #pragma unroll
    for (int n = 0; n < 16; ++n) {
        float ab = __expf(-__expf(A_log[d * 16 + n]) * dt);
        Ab[n]  = fminf(1.f - 1e-8f, fmaxf(1e-8f, ab));
        Bd[n]  = Bm[d * 16 + n] * dt;
        Cc[n]  = Cm[d * 16 + n];
        hst[n] = 0.f;
    }
    for (int s = 0; s < LSEQ; ++s) {
        int t = (dir > 0) ? s : (LSEQ - 1 - s);
        size_t row = ((size_t)b * LSEQ + t) * 256;
        float u = xz[row + d];
        float z = xz[row + DM + d];
        float acc = 0.f;
        #pragma unroll
        for (int n = 0; n < 16; ++n) {
            hst[n] = Ab[n] * hst[n] + Bd[n] * u;
            acc += Cc[n] * hst[n];
        }
        y[((size_t)b * LSEQ + t) * DM + d] = (_Float16)(acc * (z * sigmoid_f(z)));
    }
}

// ---------------------------------------------------------------------------
// Patch embed im2col: x(8,3,256,256) -> Xp f16 [8192 x 192]
// ---------------------------------------------------------------------------
__global__ __launch_bounds__(256) void patch_im2col_kernel(
    const float* __restrict__ x, _Float16* __restrict__ Xp)
{
    int idx = blockIdx.x * 256 + threadIdx.x;
    if (idx >= MROWS * 192) return;
    int m = idx / 192, k = idx % 192;
    int b = m >> 10, l = m & 1023, gy = l >> 5, gx = l & 31;
    int c = k >> 6, r = k & 63, py = r >> 3, px = r & 7;
    Xp[idx] = (_Float16)x[(((size_t)(b * 3 + c) * 256 + (gy * 8 + py)) * 256) + gx * 8 + px];
}

// ---------------------------------------------------------------------------
// Sharp conv im2col: hrec[8192 x 128] (tokens) -> Xs f16 [8192 x 1152]
// col k = (c*3+ky)*3+kx, 'SAME' zero pad on the 32x32 token grid.
// ---------------------------------------------------------------------------
__global__ __launch_bounds__(256) void sharp_im2col_kernel(
    const float* __restrict__ hrec, _Float16* __restrict__ Xs)
{
    size_t idx = (size_t)blockIdx.x * 256 + threadIdx.x;
    if (idx >= (size_t)MROWS * 1152) return;
    int m = (int)(idx / 1152), k = (int)(idx % 1152);
    int c = k / 9, q = k % 9, ky = q / 3, kx = q % 3;
    int b = m >> 10, l = m & 1023, y = l >> 5, xx = l & 31;
    int yy = y + ky - 1, xc = xx + kx - 1;
    float v = 0.f;
    if (yy >= 0 && yy < 32 && xc >= 0 && xc < 32)
        v = hrec[((size_t)(b * LSEQ + yy * 32 + xc)) * DM + c];
    Xs[idx] = (_Float16)v;
}

// pixel shuffle (r=8) + exact GELU: spre[8192 x 256] -> shimg[8,4,256,256]
__global__ __launch_bounds__(256) void pshuffle_gelu_kernel(
    const float* __restrict__ spre, float* __restrict__ shimg)
{
    int idx = blockIdx.x * 256 + threadIdx.x;
    if (idx >= 8 * 4 * 256 * 256) return;
    int X = idx & 255, Y = (idx >> 8) & 255, c = (idx >> 16) & 3, b = idx >> 18;
    int y = Y >> 3, ry = Y & 7, x = X >> 3, rx = X & 7;
    int oc = c * 64 + ry * 8 + rx;
    int m  = b * LSEQ + y * 32 + x;
    shimg[idx] = gelu_f(spre[(size_t)m * 256 + oc]);
}

// 4-tap bicubic weights (A=-0.75), matching the reference matrix
__device__ __forceinline__ void bicubic_taps(int i, int insz, int outsz,
                                             int* j, float* w)
{
    const float Ac = -0.75f;
    float scale = (float)insz / (float)outsz;
    float src = (i + 0.5f) * scale - 0.5f;
    float f = floorf(src);
    float t = src - f;
    int fi = (int)f;
    #pragma unroll
    for (int k = 0; k < 4; ++k) {
        float tt = fabsf(t - (float)(k - 1));
        float wv;
        if (tt <= 1.f)      wv = ((Ac + 2.f) * tt - (Ac + 3.f)) * tt * tt + 1.f;
        else if (tt < 2.f)  wv = Ac * (((tt - 5.f) * tt + 8.f) * tt - 4.f);
        else                wv = 0.f;
        int jj = fi + k - 1;
        j[k] = jj < 0 ? 0 : (jj > insz - 1 ? insz - 1 : jj);
        w[k] = wv;
    }
}

// vertical bicubic: sm1[b][Y][gx][c] = sum_j Mv[Y,j] * hrec[b, j*32+gx, c]
__global__ __launch_bounds__(256) void bicubic_rows_kernel(
    const float* __restrict__ hrec, float* __restrict__ sm1)
{
    int idx = blockIdx.x * 256 + threadIdx.x;
    if (idx >= 8 * 256 * 32 * 128) return;
    int c = idx & 127, i = (idx >> 7) & 31, Y = (idx >> 12) & 255, b = idx >> 20;
    int j[4]; float w[4];
    bicubic_taps(Y, 32, 256, j, w);
    float acc = 0.f;
    #pragma unroll
    for (int k = 0; k < 4; ++k)
        acc += w[k] * hrec[((size_t)(b * LSEQ + j[k] * 32 + i)) * DM + c];
    sm1[idx] = acc;
}

// horizontal bicubic fused with the smooth 3x3 conv (128ch -> 4ch) + GELU.
// one block (128 threads = channels) per output pixel, LDS tree reduce.
__global__ __launch_bounds__(128) void smooth_conv_kernel(
    const float* __restrict__ sm1, const float* __restrict__ w,
    const float* __restrict__ bias, float* __restrict__ smimg)
{
    __shared__ float red[128];
    int p = blockIdx.x;
    int X = p & 255, Y = (p >> 8) & 255, b = p >> 16;
    int c = threadIdx.x;

    int jh[3][4]; float wh[3][4]; bool vx[3];
    #pragma unroll
    for (int kx = 0; kx < 3; ++kx) {
        int Xk = X + kx - 1;
        vx[kx] = (Xk >= 0 && Xk < 256);
        if (vx[kx]) bicubic_taps(Xk, 32, 256, jh[kx], wh[kx]);
    }

    float acc[4] = {0.f, 0.f, 0.f, 0.f};
    #pragma unroll
    for (int ky = 0; ky < 3; ++ky) {
        int Yk = Y + ky - 1;
        if (Yk < 0 || Yk >= 256) continue;
        size_t base = ((size_t)(b * 256 + Yk)) * 32 * 128 + c;
        #pragma unroll
        for (int kx = 0; kx < 3; ++kx) {
            if (!vx[kx]) continue;
            float sv = 0.f;
            #pragma unroll
            for (int t = 0; t < 4; ++t)
                sv += wh[kx][t] * sm1[base + (size_t)jh[kx][t] * 128];
            #pragma unroll
            for (int oc = 0; oc < 4; ++oc)
                acc[oc] += w[((oc * 128 + c) * 3 + ky) * 3 + kx] * sv;
        }
    }
    #pragma unroll
    for (int oc = 0; oc < 4; ++oc) {
        red[c] = acc[oc]; __syncthreads();
        #pragma unroll
        for (int s = 64; s > 0; s >>= 1) { if (c < s) red[c] += red[c + s]; __syncthreads(); }
        if (c == 0)
            smimg[(((size_t)(b * 4 + oc)) * 256 + Y) * 256 + X] = gelu_f(red[0] + bias[oc]);
        __syncthreads();
    }
}

// global mean over tokens + atmosphere MLP (128 -> gelu 64 -> sigmoid 3)
__global__ __launch_bounds__(128) void mean_atm_kernel(
    const float* __restrict__ feat, const float* __restrict__ w1,
    const float* __restrict__ b1, const float* __restrict__ w2,
    const float* __restrict__ b2, float* __restrict__ catm)
{
    __shared__ float g[128];
    __shared__ float a1[64];
    int b = blockIdx.x, d = threadIdx.x;
    float s = 0.f;
    for (int l = 0; l < LSEQ; ++l)
        s += feat[((size_t)b * LSEQ + l) * DM + d];
    g[d] = s * (1.f / LSEQ);
    __syncthreads();
    if (d < 64) {
        float s2 = b1[d];
        for (int i = 0; i < 128; ++i) s2 += g[i] * w1[i * 64 + d];
        a1[d] = gelu_f(s2);
    }
    __syncthreads();
    if (d < 3) {
        float s3 = b2[d];
        for (int i = 0; i < 64; ++i) s3 += a1[i] * w2[i * 3 + d];
        catm[b * 3 + d] = sigmoid_f(s3);
    }
}

// fuse 3x3 conv (8ch -> 4ch) + dehaze equation + clip -> d_out (8,3,256,256)
__global__ __launch_bounds__(256) void fuse_dehaze_kernel(
    const float* __restrict__ shimg, const float* __restrict__ smimg,
    const float* __restrict__ fw, const float* __restrict__ fb,
    const float* __restrict__ x, const float* __restrict__ catm,
    float* __restrict__ out)
{
    int p = blockIdx.x * 256 + threadIdx.x;
    if (p >= 8 * 256 * 256) return;
    int X = p & 255, Y = (p >> 8) & 255, b = p >> 16;

    float m[4] = {fb[0], fb[1], fb[2], fb[3]};
    #pragma unroll
    for (int ic = 0; ic < 8; ++ic) {
        const float* src = (ic < 4) ? shimg : smimg;
        int ch = ic & 3;
        size_t base = ((size_t)(b * 4 + ch)) * 256 * 256;
        #pragma unroll
        for (int ky = 0; ky < 3; ++ky) {
            int Yk = Y + ky - 1;
            if (Yk < 0 || Yk >= 256) continue;
            #pragma unroll
            for (int kx = 0; kx < 3; ++kx) {
                int Xk = X + kx - 1;
                if (Xk < 0 || Xk >= 256) continue;
                float v = src[base + (size_t)Yk * 256 + Xk];
                #pragma unroll
                for (int oc = 0; oc < 4; ++oc)
                    m[oc] += fw[((oc * 8 + ic) * 3 + ky) * 3 + kx] * v;
            }
        }
    }
    float Kv = sigmoid_f(m[0]) * 10.f;
    #pragma unroll
    for (int c = 0; c < 3; ++c) {
        float ca = catm[b * 3 + c];
        float xv = x[(((size_t)(b * 3 + c)) * 256 + Y) * 256 + X];
        float J = Kv * xv - Kv * ca + ca + m[1 + c];
        out[(((size_t)(b * 3 + c)) * 256 + Y) * 256 + X] = fminf(1.f, fmaxf(0.f, J));
    }
}

// ---------------------------------------------------------------------------
// Host side
// ---------------------------------------------------------------------------
template <int EPI, bool CH>
static void gemm_launch(const _Float16* A, int lda, const _Float16* Bp,
                        const float* bias, const float* res, int ldr,
                        const float* pos, void* C, int ldc, int N, int K,
                        hipStream_t s)
{
    dim3 grid(MROWS / 64, N / 32);
    gemm_wmma_kernel<EPI, CH><<<grid, 256, 0, s>>>(A, lda, Bp, bias, res,
                                                   ldr, pos, C, ldc, K);
}

static void pack_wt(const float* W, _Float16* out, int K, int N, hipStream_t s)
{
    pack_wt_kernel<<<(K * N + 255) / 256, 256, 0, s>>>(W, out, K, N);
}

struct S4P {
    const float *A_log, *B, *C, *log_dt, *in_w, *in_b, *out_w, *out_b, *ln_g, *ln_b;
};
struct BlkP {
    S4P fwd, bwd;
    const float *fus_w, *fus_b, *ln_g, *ln_b;
};

extern "C" void kernel_launch(void* const* d_in, const int* in_sizes, int n_in,
                              void* d_out, int out_size, void* d_ws, size_t ws_size,
                              hipStream_t stream)
{
    (void)in_sizes; (void)n_in; (void)out_size; (void)ws_size;
    int i = 0;
    auto nxt = [&]() { return (const float*)d_in[i++]; };

    const float* x       = nxt();
    const float* patch_w = nxt();
    const float* patch_b = nxt();
    const float* pos     = nxt();
    BlkP blk[6];
    auto read_s4 = [&](S4P& s) {
        s.A_log = nxt(); s.B = nxt(); s.C = nxt(); s.log_dt = nxt();
        s.in_w = nxt(); s.in_b = nxt(); s.out_w = nxt(); s.out_b = nxt();
        s.ln_g = nxt(); s.ln_b = nxt();
    };
    for (int b = 0; b < 6; ++b) {
        read_s4(blk[b].fwd);
        read_s4(blk[b].bwd);
        blk[b].fus_w = nxt(); blk[b].fus_b = nxt();
        blk[b].ln_g  = nxt(); blk[b].ln_b  = nxt();
    }
    const float* rec_w    = nxt();
    const float* rec_b    = nxt();
    const float* sharp_w  = nxt();
    const float* sharp_b  = nxt();
    const float* smooth_w = nxt();
    const float* smooth_b = nxt();
    const float* fuse_w   = nxt();
    const float* fuse_b   = nxt();
    const float* atm1_w   = nxt();
    const float* atm1_b   = nxt();
    const float* atm2_w   = nxt();
    const float* atm2_b   = nxt();

    // ---- workspace layout --------------------------------------------------
    float* ws    = (float*)d_ws;
    float* feat  = ws;                                   // 8192*128
    float* hbuf  = feat  + (size_t)MROWS * DM;           // 8192*128 (fusion out)
    float* xz    = hbuf  + (size_t)MROWS * DM;           // 8192*256
    float* hrec  = xz    + (size_t)MROWS * 256;          // 8192*128
    float* spre  = hrec  + (size_t)MROWS * DM;           // 8192*256
    float* shimg = spre  + (size_t)MROWS * 256;          // 8*4*256*256
    float* sm1   = shimg + (size_t)8 * 4 * 256 * 256;    // 8*256*32*128
    float* smimg = sm1   + (size_t)8 * 256 * 32 * 128;   // 8*4*256*256
    float* catm  = smimg + (size_t)8 * 4 * 256 * 256;    // 24 (+pad)

    // f16 region (16B aligned: all previous sizes are multiples of 8 floats)
    _Float16* feat_h = (_Float16*)(catm + 32);           // 8192*128
    _Float16* lnh    = feat_h + (size_t)MROWS * DM;      // 8192*128
    _Float16* ybuf_h = lnh    + (size_t)MROWS * DM;      // 8192*128
    _Float16* cat_h  = ybuf_h + (size_t)MROWS * DM;      // 8192*256
    _Float16* Ahalf  = cat_h  + (size_t)MROWS * 256;     // 8192*1152

    // f16 packed-weight arena (all [N][K] row-major)
    _Float16* wp = Ahalf + (size_t)MROWS * 1152;
    _Float16* patch_wt = wp;                 wp += 128 * 192;
    _Float16* in_wt[6][2];
    _Float16* out_wt[6][2];
    _Float16* fus_wt[6];
    for (int b = 0; b < 6; ++b) {
        for (int d = 0; d < 2; ++d) { in_wt[b][d]  = wp; wp += 256 * 128; }
        for (int d = 0; d < 2; ++d) { out_wt[b][d] = wp; wp += 128 * 128; }
        fus_wt[b] = wp; wp += 128 * 256;
    }
    _Float16* rec_wt   = wp; wp += 128 * 128;
    _Float16* sharp_wt = wp; wp += 256 * 1152;

    // ---- 0. pack all GEMM weights to f16 [N][K] ---------------------------
    pack_wt(patch_w, patch_wt, 192, 128, stream);
    for (int b = 0; b < 6; ++b) {
        pack_wt(blk[b].fwd.in_w,  in_wt[b][0],  128, 256, stream);
        pack_wt(blk[b].bwd.in_w,  in_wt[b][1],  128, 256, stream);
        pack_wt(blk[b].fwd.out_w, out_wt[b][0], 128, 128, stream);
        pack_wt(blk[b].bwd.out_w, out_wt[b][1], 128, 128, stream);
        pack_wt(blk[b].fus_w,     fus_wt[b],    256, 128, stream);
    }
    pack_wt(rec_w, rec_wt, 128, 128, stream);
    pack_direct_kernel<<<(256 * 1152 + 255) / 256, 256, 0, stream>>>(
        sharp_w, sharp_wt, 256 * 1152);   // sharp_w is already [N][K] flat

    // ---- 1. patch embedding (WMMA GEMM, K=192) ----------------------------
    patch_im2col_kernel<<<(MROWS * 192) / 256, 256, 0, stream>>>(x, Ahalf);
    gemm_launch<EPI_BIAS_POS, false>(Ahalf, 192, patch_wt, patch_b,
                                     nullptr, 0, pos, feat, DM, DM, 192, stream);

    // ---- 2. six bidirectional Vim blocks ----------------------------------
    for (int b = 0; b < 6; ++b) {
        for (int dir = 0; dir < 2; ++dir) {
            const S4P& s = dir == 0 ? blk[b].fwd : blk[b].bwd;
            ln_kernel<<<MROWS, 128, 0, stream>>>(feat, s.ln_g, s.ln_b,
                                                 nullptr, lnh);
            gemm_launch<EPI_BIAS, false>(lnh, DM, in_wt[b][dir], s.in_b,
                                         nullptr, 0, nullptr, xz, 256, 256, DM, stream);
            scan_kernel<<<8, 128, 0, stream>>>(xz, s.A_log, s.B, s.C, s.log_dt,
                                               ybuf_h, dir == 0 ? 1 : -1);
            gemm_launch<EPI_BIAS_RES_CLIP, true>(ybuf_h, DM, out_wt[b][dir], s.out_b,
                                                 feat, DM, nullptr,
                                                 cat_h + dir * DM, 256, DM, DM, stream);
        }
        gemm_launch<EPI_BIAS, false>(cat_h, 256, fus_wt[b], blk[b].fus_b,
                                     nullptr, 0, nullptr, hbuf, DM, DM, 256, stream);
        ln_kernel<<<MROWS, 128, 0, stream>>>(hbuf, blk[b].ln_g, blk[b].ln_b,
                                             feat, feat_h);
    }

    // ---- 3. reconstruction projection (A = f16 feat) ----------------------
    gemm_launch<EPI_BIAS, false>(feat_h, DM, rec_wt, rec_b,
                                 nullptr, 0, nullptr, hrec, DM, DM, DM, stream);

    // ---- 4. sharp path: implicit-GEMM 3x3 conv (K=1152) + shuffle + GELU --
    sharp_im2col_kernel<<<(int)(((size_t)MROWS * 1152) / 256), 256, 0, stream>>>(hrec, Ahalf);
    gemm_launch<EPI_BIAS, false>(Ahalf, 1152, sharp_wt, sharp_b,
                                 nullptr, 0, nullptr, spre, 256, 256, 1152, stream);
    pshuffle_gelu_kernel<<<(8 * 4 * 256 * 256) / 256, 256, 0, stream>>>(spre, shimg);

    // ---- 5. smooth path: separable bicubic fused with 3x3 conv ------------
    bicubic_rows_kernel<<<(8 * 256 * 32 * 128) / 256, 256, 0, stream>>>(hrec, sm1);
    smooth_conv_kernel<<<8 * 256 * 256, 128, 0, stream>>>(sm1, smooth_w, smooth_b, smimg);

    // ---- 6. atmosphere estimate -------------------------------------------
    mean_atm_kernel<<<8, 128, 0, stream>>>(feat, atm1_w, atm1_b, atm2_w, atm2_b, catm);

    // ---- 7. fuse conv + dehaze equation -----------------------------------
    fuse_dehaze_kernel<<<(8 * 256 * 256) / 256, 256, 0, stream>>>(
        shimg, smimg, fuse_w, fuse_b, x, catm, (float*)d_out);
}
